// GWNBlock_55173149884654
// MI455X (gfx1250) — matrix operand order repaired and checked by reading, not compiled, not run on hardware
//
#include <hip/hip_runtime.h>
#include <cstddef>

// ---------------- problem constants ----------------
#define BB   16
#define NN   1024
#define CC   64
#define TT   24
#define CT   1536            // C*T
#define NODES 16384          // B*N
#define EMB  10
#define OUTELEMS ((size_t)BB*NN*CC*TT)   // 25,165,824

typedef __attribute__((ext_vector_type(16))) __bf16 v16bf;
typedef __attribute__((ext_vector_type(8)))  float  v8f;
typedef __attribute__((ext_vector_type(4)))  unsigned int v4u;

union AFu { v16bf v; unsigned int w[8]; v4u q[2]; };

__device__ __forceinline__ unsigned short f2bf(float f) {
    union { float f; unsigned int u; } c; c.f = f;
    unsigned int u = c.u;
    u += 0x7FFFu + ((u >> 16) & 1u);   // round-to-nearest-even
    return (unsigned short)(u >> 16);
}
__device__ __forceinline__ unsigned int packbf(float lo, float hi) {
    return (unsigned int)f2bf(lo) | ((unsigned int)f2bf(hi) << 16);
}
__device__ __forceinline__ v8f zero8() {
    v8f z;
    #pragma unroll
    for (int i = 0; i < 8; ++i) z[i] = 0.f;
    return z;
}

#define WMMA_BF16(A, B, Cacc) \
    __builtin_amdgcn_wmma_f32_16x16x32_bf16(false, (A), false, (B), (short)0, (Cacc), false, false)

// CDNA5 async global->LDS copy of 16 bytes (per enabled lane), ASYNCcnt-tracked.
__device__ __forceinline__ void async_g2l_b128(unsigned lds_off, unsigned long long gaddr) {
    asm volatile("global_load_async_to_lds_b128 %0, %1, off"
                 :: "v"(lds_off), "v"(gaddr) : "memory");
}
__device__ __forceinline__ void wait_async0() {
    asm volatile("s_wait_asynccnt 0" ::: "memory");
}

// ---------------------------------------------------------------------------
// Kernel: adj = softmax(relu(e1 @ e2^T)) per row, bf16 [N,N] (K-pairs contiguous)
// ---------------------------------------------------------------------------
__global__ __launch_bounds__(256) void k_adj(const float* __restrict__ e1,
                                             const float* __restrict__ e2,
                                             unsigned short* __restrict__ adj_bf)
{
    __shared__ float row[NN];
    __shared__ float red[256];
    const int m = blockIdx.x;
    float a[EMB];
    #pragma unroll
    for (int k = 0; k < EMB; ++k) a[k] = e1[m * EMB + k];

    for (int n = threadIdx.x; n < NN; n += 256) {
        float d = 0.f;
        #pragma unroll
        for (int k = 0; k < EMB; ++k) d += a[k] * e2[n * EMB + k];
        row[n] = d > 0.f ? d : 0.f;
    }
    __syncthreads();

    float mx = -1e30f;
    for (int n = threadIdx.x; n < NN; n += 256) mx = fmaxf(mx, row[n]);
    red[threadIdx.x] = mx; __syncthreads();
    for (int s = 128; s > 0; s >>= 1) {
        if (threadIdx.x < s) red[threadIdx.x] = fmaxf(red[threadIdx.x], red[threadIdx.x + s]);
        __syncthreads();
    }
    mx = red[0]; __syncthreads();

    float sm = 0.f;
    for (int n = threadIdx.x; n < NN; n += 256) {
        float e = __expf(row[n] - mx);
        row[n] = e; sm += e;
    }
    red[threadIdx.x] = sm; __syncthreads();
    for (int s = 128; s > 0; s >>= 1) {
        if (threadIdx.x < s) red[threadIdx.x] += red[threadIdx.x + s];
        __syncthreads();
    }
    const float inv = 1.f / red[0];
    for (int n = threadIdx.x; n < NN; n += 256)
        adj_bf[(size_t)m * NN + n] = f2bf(row[n] * inv);
}

// ---------------------------------------------------------------------------
// Kernel: pre-pack all GEMM weights into fragment-ordered uint32 buffers.
// ---------------------------------------------------------------------------
__global__ __launch_bounds__(256) void k_wpack(const float* __restrict__ w_tanh,
                                               const float* __restrict__ w_sig,
                                               const float* __restrict__ w_lin,
                                               const float* __restrict__ w_res,
                                               unsigned int* __restrict__ wf_tcn,
                                               unsigned int* __restrict__ wf_lin,
                                               unsigned int* __restrict__ wf_res)
{
    const int idx = blockIdx.x * 256 + threadIdx.x;     // 0..16383
    const int r    = idx & 7;
    const int lane = (idx >> 3) & 31;
    const int lh = lane >> 4, lm = lane & 15;
    const int kpart = (lh << 4) + (r << 1);

    if (idx < 8192) {                 // gated TCN weights (tap-folded K=128)
        const int sub = idx;
        const int mat = sub >> 12;
        const int rem = sub & 4095;
        const int mt = rem >> 10, ks = (rem >> 8) & 3;
        const int k0 = ks * 32 + kpart;
        const int m  = mt * 16 + lm;
        const int off = (k0 < 64) ? (m * 128 + k0 * 2)
                                  : (m * 128 + (k0 - 64) * 2 + 1);
        const float* src = mat ? w_sig : w_tanh;
        wf_tcn[sub] = packbf(src[off], src[off + 2]);
    } else if (idx < 14336) {         // linear weights K=192
        const int sub = idx - 8192;
        const int ksmt = sub >> 8;    // mt*6+ks
        const int ks = ksmt % 6, mt = ksmt / 6;
        const int k0 = ks * 32 + kpart;
        const int off = (mt * 16 + lm) * 192 + k0;
        wf_lin[sub] = packbf(w_lin[off], w_lin[off + 1]);
    } else {                          // residual weights K=64
        const int sub = idx - 14336;
        const int ksmt = sub >> 8;    // mt*2+ks
        const int ks = ksmt & 1, mt = ksmt >> 1;
        const int k0 = ks * 32 + kpart;
        const int off = (mt * 16 + lm) * 64 + k0;
        wf_res[sub] = packbf(w_res[off], w_res[off + 1]);
    }
}

// ---------------------------------------------------------------------------
// Kernel: pack x into channel-pair-interleaved bf16:
//   xp_u32[g*768 + (c>>1)*24 + t] = pack(x[g,c,t], x[g,c+1,t])
// ---------------------------------------------------------------------------
__global__ __launch_bounds__(256) void k_xpack(const float* __restrict__ x,
                                               unsigned int* __restrict__ xp)
{
    const size_t p0 = ((size_t)blockIdx.x * 256 + threadIdx.x) * 4;
    #pragma unroll
    for (int i = 0; i < 4; ++i) {
        const size_t q = p0 + i;                  // < NODES*768
        const size_t g = q / 768;
        const int rem = (int)(q % 768);
        const int kp = rem / 24, t = rem % 24;
        const float* b = x + g * CT + kp * 48 + t;
        xp[q] = packbf(b[0], b[24]);
    }
}

// ---------------------------------------------------------------------------
// Kernel: gated TCN.  D[64,24] = tanh(Wt·X)*sig(Ws·X), X = tap-concat (K=128).
// ---------------------------------------------------------------------------
__global__ __launch_bounds__(256) void k_tcn(const unsigned int* __restrict__ xp,
                                             const unsigned int* __restrict__ wf_tcn,
                                             const float* __restrict__ b_tanh,
                                             const float* __restrict__ b_sig,
                                             unsigned short* __restrict__ h_bf)
{
    __shared__ __align__(16) unsigned int bm[2][32][68];
    const int tid = threadIdx.x;
    const int node0 = blockIdx.x * 2;
    {
        const int row = tid >> 1;             // 0..127 = (nd, kp)
        const int fnd = row >> 6, kp = row & 63;
        const int th = (tid & 1) * 16;
        const size_t base = (size_t)(node0 + fnd) * 768 + (kp & 31) * 24;
        #pragma unroll
        for (int i = 0; i < 16; ++i) {
            const int t = th + i;
            const int tsrc = (kp < 32) ? t - 1 : t;   // tap0 causal shift
            unsigned int v = 0;
            if (t < TT && tsrc >= 0 && tsrc < TT) v = xp[base + tsrc];
            bm[fnd][t][kp] = v;
        }
    }
    __syncthreads();

    const int lane = tid & 31, w = tid >> 5;
    const int nd = w >> 2, mt = w & 3;
    const int g = node0 + nd;
    const int lm = lane & 15, lh = lane >> 4;

    v8f accT[2], accS[2];
    accT[0] = zero8(); accT[1] = zero8(); accS[0] = zero8(); accS[1] = zero8();

    for (int ks = 0; ks < 4; ++ks) {
        AFu aT, aS;
        const v4u* wt  = (const v4u*)(wf_tcn + (((mt * 4 + ks) * 32 + lane) << 3));
        const v4u* wsg = (const v4u*)(wf_tcn + 4096 + (((mt * 4 + ks) * 32 + lane) << 3));
        aT.q[0] = wt[0];  aT.q[1] = wt[1];
        aS.q[0] = wsg[0]; aS.q[1] = wsg[1];
        #pragma unroll
        for (int nt = 0; nt < 2; ++nt) {
            AFu bf;
            const unsigned int* bp = &bm[nd][nt * 16 + lm][ks * 16 + (lh << 3)];
            bf.q[0] = *(const v4u*)bp;
            bf.q[1] = *(const v4u*)(bp + 4);
            accT[nt] = WMMA_BF16(aT.v, bf.v, accT[nt]);
            accS[nt] = WMMA_BF16(aS.v, bf.v, accS[nt]);
        }
    }

    #pragma unroll
    for (int nt = 0; nt < 2; ++nt) {
        const int t = nt * 16 + lm;
        if (t < TT) {
            #pragma unroll
            for (int r = 0; r < 8; ++r) {
                const int o = mt * 16 + r + (lh << 3);
                float tv = tanhf(accT[nt][r] + b_tanh[o]);
                float sv = accS[nt][r] + b_sig[o];
                sv = 1.f / (1.f + __expf(-sv));
                h_bf[((size_t)(g >> 1) * CT + o * TT + t) * 2 + (g & 1)] = f2bf(tv * sv);
            }
        }
    }
}

// ---------------------------------------------------------------------------
// Kernel: diffusion GEMM  Out[b,m,ct] = sum_n adj[m,n]*In[b,n,ct]
// In/Out node-pair-interleaved uint32: [b*512+np][1536].
// Block tile 128x64, K=64 per stage, double-buffered LDS;
// A tile filled with async global->LDS b128, B tile pair-transposed via VGPRs.
// ---------------------------------------------------------------------------
__global__ __launch_bounds__(256) void k_diff(const unsigned int* __restrict__ adj_u32,
                                              const unsigned int* __restrict__ In,
                                              unsigned int* __restrict__ Out)
{
    __shared__ __align__(16) unsigned int Ap[2][128][36];
    __shared__ __align__(16) unsigned int Bp[2][64][36];
    const int tid = threadIdx.x;
    const int ct0 = blockIdx.x * 64;
    const int m0  = blockIdx.y * 128;
    const int b   = blockIdx.z;
    const int lane = tid & 31, w = tid >> 5;
    const int mw = w >> 1, nw = w & 1;
    const int lm = lane & 15, lh = lane >> 4;
    const size_t inb = (size_t)b * 512 * CT;

    v8f acc[2][2];
    #pragma unroll
    for (int i = 0; i < 2; ++i) { acc[i][0] = zero8(); acc[i][1] = zero8(); }

    const int a_m  = tid >> 1, a_h = (tid & 1) << 4;  // A fill: row, 16-dword half
    const int b_kp = tid >> 3, b_c = (tid & 7) << 3;  // B fill: kpair-row, 8-col chunk

    const unsigned int* arow = adj_u32 + (size_t)(m0 + a_m) * 512 + a_h;
    const unsigned int* brow = In + inb + (size_t)b_kp * CT + ct0 + b_c;

    auto fillA = [&](int buf, int ks) {
        const unsigned lds0 = (unsigned)(size_t)&Ap[buf][a_m][a_h];
        const unsigned long long g0 = (unsigned long long)(size_t)(arow + ks * 32);
        #pragma unroll
        for (int i = 0; i < 4; ++i)
            async_g2l_b128(lds0 + 16u * i, g0 + 16ull * i);
    };
    auto fillB = [&](int buf, int ks) {
        const unsigned int* bs = brow + (size_t)(ks * 32) * CT;
        v4u t0 = *(const v4u*)bs;
        v4u t1 = *(const v4u*)(bs + 4);
        #pragma unroll
        for (int i = 0; i < 4; ++i) Bp[buf][b_c + i][b_kp] = t0[i];
        #pragma unroll
        for (int i = 0; i < 4; ++i) Bp[buf][b_c + 4 + i][b_kp] = t1[i];
    };

    fillA(0, 0); fillB(0, 0);
    wait_async0();
    __syncthreads();

    for (int ks = 0; ks < 16; ++ks) {
        const int cur = ks & 1;
        if (ks + 1 < 16) {
            fillA(cur ^ 1, ks + 1);
            fillB(cur ^ 1, ks + 1);
            if (ks + 2 < 16)
                __builtin_prefetch(brow + (size_t)((ks + 2) * 32) * CT, 0, 1);
        }

        #pragma unroll
        for (int sub = 0; sub < 2; ++sub) {
            AFu af[2], bfr[2];
            #pragma unroll
            for (int mi = 0; mi < 2; ++mi) {
                const unsigned int* ap = &Ap[cur][mw * 32 + mi * 16 + lm][sub * 16 + (lh << 3)];
                af[mi].q[0] = *(const v4u*)ap;
                af[mi].q[1] = *(const v4u*)(ap + 4);
            }
            #pragma unroll
            for (int ni = 0; ni < 2; ++ni) {
                const unsigned int* bp = &Bp[cur][nw * 32 + ni * 16 + lm][sub * 16 + (lh << 3)];
                bfr[ni].q[0] = *(const v4u*)bp;
                bfr[ni].q[1] = *(const v4u*)(bp + 4);
            }
            #pragma unroll
            for (int mi = 0; mi < 2; ++mi)
                #pragma unroll
                for (int ni = 0; ni < 2; ++ni)
                    acc[mi][ni] = WMMA_BF16(af[mi].v, bfr[ni].v, acc[mi][ni]);
        }

        if (ks + 1 < 16) {
            wait_async0();          // our async fills of the next buffer are in LDS
            __syncthreads();        // everyone done reading cur, done writing nxt
        }
    }

    #pragma unroll
    for (int mi = 0; mi < 2; ++mi)
        #pragma unroll
        for (int ni = 0; ni < 2; ++ni) {
            const int mbase = m0 + mw * 32 + mi * 16 + (lh << 3);    // even
            const int ct = ct0 + nw * 32 + ni * 16 + lm;
            #pragma unroll
            for (int rp = 0; rp < 4; ++rp) {
                const unsigned int v = packbf(acc[mi][ni][2 * rp], acc[mi][ni][2 * rp + 1]);
                Out[inb + (size_t)((mbase >> 1) + rp) * CT + ct] = v;
            }
        }
}

// ---------------------------------------------------------------------------
// Kernel: linear over concat supports.  lin = Wlin[64,192]·Cat + b  (fp32 out)
// ---------------------------------------------------------------------------
__global__ __launch_bounds__(256) void k_lin(const unsigned short* __restrict__ h_bf,
                                             const unsigned short* __restrict__ s1,
                                             const unsigned short* __restrict__ s2,
                                             const unsigned int* __restrict__ wf_lin,
                                             const float* __restrict__ b_lin,
                                             float* __restrict__ lin_out)
{
    __shared__ __align__(16) unsigned int cm[2][32][100];
    const int tid = threadIdx.x;
    const int node0 = blockIdx.x * 2;

    for (int idx = tid; idx < 6144; idx += 256) {
        const int fnd = idx / 3072, rem = idx % 3072;
        const int kp = rem >> 5, t = rem & 31;
        unsigned int v = 0;
        if (t < TT) {
            const int q = kp >> 5;
            const int c = (kp & 31) * 2;
            const unsigned short* src = (q == 0) ? h_bf : (q == 1) ? s1 : s2;
            const int g = node0 + fnd;
            const size_t base = ((size_t)(g >> 1) * CT) * 2 + (g & 1);
            const unsigned int lo = src[base + (size_t)(c * TT + t) * 2];
            const unsigned int hi = src[base + (size_t)((c + 1) * TT + t) * 2];
            v = lo | (hi << 16);
        }
        cm[fnd][t][kp] = v;
    }
    __syncthreads();

    const int lane = tid & 31, w = tid >> 5;
    const int nd = w >> 2, mt = w & 3;
    const int g = node0 + nd;
    const int lm = lane & 15, lh = lane >> 4;

    v8f acc[2]; acc[0] = zero8(); acc[1] = zero8();

    for (int ks = 0; ks < 6; ++ks) {
        AFu af;
        const v4u* wp = (const v4u*)(wf_lin + (((mt * 6 + ks) * 32 + lane) << 3));
        af.q[0] = wp[0]; af.q[1] = wp[1];
        #pragma unroll
        for (int nt = 0; nt < 2; ++nt) {
            AFu bf;
            const unsigned int* bp = &cm[nd][nt * 16 + lm][ks * 16 + (lh << 3)];
            bf.q[0] = *(const v4u*)bp;
            bf.q[1] = *(const v4u*)(bp + 4);
            acc[nt] = WMMA_BF16(af.v, bf.v, acc[nt]);
        }
    }

    #pragma unroll
    for (int nt = 0; nt < 2; ++nt) {
        const int t = nt * 16 + lm;
        if (t < TT) {
            #pragma unroll
            for (int r = 0; r < 8; ++r) {
                const int o = mt * 16 + r + (lh << 3);
                lin_out[(size_t)g * CT + o * TT + t] = acc[nt][r] + b_lin[o];
            }
        }
    }
}

// ---------------------------------------------------------------------------
// Kernel: BN batch stats per (c,t)
// ---------------------------------------------------------------------------
__global__ __launch_bounds__(256) void k_stats(const float* __restrict__ lin,
                                               float* __restrict__ mean,
                                               float* __restrict__ invstd)
{
    __shared__ float rs[256], rq[256];
    const int ct = blockIdx.x;
    float s = 0.f, q = 0.f;
    for (int bn = threadIdx.x; bn < NODES; bn += 256) {
        const float v = lin[(size_t)bn * CT + ct];
        s += v; q += v * v;
    }
    rs[threadIdx.x] = s; rq[threadIdx.x] = q; __syncthreads();
    for (int st = 128; st > 0; st >>= 1) {
        if (threadIdx.x < st) { rs[threadIdx.x] += rs[threadIdx.x + st]; rq[threadIdx.x] += rq[threadIdx.x + st]; }
        __syncthreads();
    }
    if (threadIdx.x == 0) {
        const float mn = rs[0] * (1.f / NODES);
        const float vr = rq[0] * (1.f / NODES) - mn * mn;
        mean[ct] = mn;
        invstd[ct] = rsqrtf(vr + 1e-5f);
    }
}

// ---------------------------------------------------------------------------
// Kernel: finalize.  skip = Wres·x + b_res (WMMA); h = relu(BN(lin));
// out0 = h + skip; out_h = h (in place over lin scratch).
// ---------------------------------------------------------------------------
__global__ __launch_bounds__(256) void k_fin(const unsigned int* __restrict__ xp,
                                             const unsigned int* __restrict__ wf_res,
                                             const float* __restrict__ b_res,
                                             const float* __restrict__ gamma,
                                             const float* __restrict__ beta,
                                             const float* __restrict__ mean,
                                             const float* __restrict__ invstd,
                                             float* __restrict__ out0,
                                             float* __restrict__ out_h)
{
    __shared__ __align__(16) unsigned int xs[2][32][36];
    const int tid = threadIdx.x;
    const int node0 = blockIdx.x * 2;

    for (int idx = tid; idx < 2048; idx += 256) {
        const int fnd = idx >> 10, rem = idx & 1023;
        const int kp = rem >> 5, t = rem & 31;
        unsigned int v = 0;
        if (t < TT) v = xp[(size_t)(node0 + fnd) * 768 + kp * 24 + t];
        xs[fnd][t][kp] = v;
    }
    __syncthreads();

    const int lane = tid & 31, w = tid >> 5;
    const int nd = w >> 2, mt = w & 3;
    const int g = node0 + nd;
    const int lm = lane & 15, lh = lane >> 4;

    v8f acc[2]; acc[0] = zero8(); acc[1] = zero8();

    for (int ks = 0; ks < 2; ++ks) {
        AFu af;
        const v4u* wp = (const v4u*)(wf_res + (((mt * 2 + ks) * 32 + lane) << 3));
        af.q[0] = wp[0]; af.q[1] = wp[1];
        #pragma unroll
        for (int nt = 0; nt < 2; ++nt) {
            AFu bf;
            const unsigned int* bp = &xs[nd][nt * 16 + lm][ks * 16 + (lh << 3)];
            bf.q[0] = *(const v4u*)bp;
            bf.q[1] = *(const v4u*)(bp + 4);
            acc[nt] = WMMA_BF16(af.v, bf.v, acc[nt]);
        }
    }

    #pragma unroll
    for (int nt = 0; nt < 2; ++nt) {
        const int t = nt * 16 + lm;
        if (t < TT) {
            #pragma unroll
            for (int r = 0; r < 8; ++r) {
                const int o = mt * 16 + r + (lh << 3);
                const size_t idx = (size_t)g * CT + o * TT + t;
                const int ct = o * TT + t;
                const float lv = out_h[idx];
                float v = gamma[o] * (lv - mean[ct]) * invstd[ct] + beta[o];
                v = fmaxf(v, 0.f);
                const float sk = acc[nt][r] + b_res[o];
                out0[idx] = v + sk;
                out_h[idx] = v;
            }
        }
    }
}

// ---------------------------------------------------------------------------
extern "C" void kernel_launch(void* const* d_in, const int* in_sizes, int n_in,
                              void* d_out, int out_size, void* d_ws, size_t ws_size,
                              hipStream_t stream)
{
    (void)in_sizes; (void)n_in; (void)out_size; (void)ws_size;

    const float* x      = (const float*)d_in[0];
    const float* w_tanh = (const float*)d_in[1];
    const float* b_tanh = (const float*)d_in[2];
    const float* w_sig  = (const float*)d_in[3];
    const float* b_sig  = (const float*)d_in[4];
    const float* e1     = (const float*)d_in[5];
    const float* e2     = (const float*)d_in[6];
    const float* w_lin  = (const float*)d_in[7];
    const float* b_lin  = (const float*)d_in[8];
    const float* gamma  = (const float*)d_in[9];
    const float* beta   = (const float*)d_in[10];
    const float* w_res  = (const float*)d_in[11];
    const float* b_res  = (const float*)d_in[12];

    float* out0  = (float*)d_out;
    float* out_h = out0 + OUTELEMS;

    // workspace layout
    char* ws = (char*)d_ws;
    const size_t ADJ_B = (size_t)NN * NN * sizeof(unsigned short);     // 2 MB
    const size_t SUP_B = (size_t)NODES * CT * sizeof(unsigned short);  // 48 MB
    unsigned short* adj_bf = (unsigned short*)ws;
    unsigned short* h_bf   = (unsigned short*)(ws + ADJ_B);
    unsigned short* s1_bf  = (unsigned short*)(ws + ADJ_B + SUP_B);
    unsigned short* s2_bf  = (unsigned short*)(ws + ADJ_B + 2 * SUP_B);
    unsigned int*   xp     = (unsigned int*)(ws + ADJ_B + 3 * SUP_B);  // 48 MB
    unsigned int*   wf_tcn = (unsigned int*)(ws + ADJ_B + 4 * SUP_B);
    unsigned int*   wf_lin = wf_tcn + 8192;
    unsigned int*   wf_res = wf_lin + 6144;
    float*          mean   = (float*)(wf_res + 2048);
    float*          invstd = mean + CT;

    k_adj  <<<NN, 256, 0, stream>>>(e1, e2, adj_bf);
    k_wpack<<<64, 256, 0, stream>>>(w_tanh, w_sig, w_lin, w_res, wf_tcn, wf_lin, wf_res);
    k_xpack<<<(NODES * 768) / 1024, 256, 0, stream>>>(x, xp);
    k_tcn  <<<NODES / 2, 256, 0, stream>>>(xp, wf_tcn, b_tanh, b_sig, h_bf);
    k_diff <<<dim3(CT / 64, NN / 128, BB), 256, 0, stream>>>((const unsigned int*)adj_bf,
                                                             (const unsigned int*)h_bf,
                                                             (unsigned int*)s1_bf);
    k_diff <<<dim3(CT / 64, NN / 128, BB), 256, 0, stream>>>((const unsigned int*)adj_bf,
                                                             (const unsigned int*)s1_bf,
                                                             (unsigned int*)s2_bf);
    k_lin  <<<NODES / 2, 256, 0, stream>>>(h_bf, s1_bf, s2_bf, wf_lin, b_lin, out_h);
    k_stats<<<CT, 256, 0, stream>>>(out_h, mean, invstd);
    k_fin  <<<NODES / 2, 256, 0, stream>>>(xp, wf_res, b_res, gamma, beta, mean, invstd, out0, out_h);
}
